// PointNet2FeaturePropagation_81733227643070
// MI455X (gfx1250) — matrix operand order repaired and checked by reading, not compiled, not run on hardware
//
#include <hip/hip_runtime.h>
#include <hip/hip_bf16.h>

// ---------------------------------------------------------------------------
// Types for WMMA / TDM (CDNA5 / gfx1250, wave32)
// ---------------------------------------------------------------------------
typedef __attribute__((ext_vector_type(16))) __bf16 v16bf;
typedef __attribute__((ext_vector_type(8)))  __bf16 v8bf;
typedef __attribute__((ext_vector_type(8)))  float  v8f;
typedef __attribute__((ext_vector_type(8)))  unsigned short v8us;
typedef unsigned int u32x4 __attribute__((ext_vector_type(4)));
typedef int          i32x4 __attribute__((ext_vector_type(4)));
typedef int          i32x8 __attribute__((ext_vector_type(8)));

// Problem constants (match setup_inputs)
#define BB   16
#define NN1  1024
#define NN2  4096
#define CC1  256
#define CC2  128
#define HH   256
#define KIN1 (CC1 + CC2)       // 384
#define NTOT (BB * NN2)        // 65536
#define BN_EPS 1e-5f
#define DIST_MIN 1e-10f

#if defined(__has_builtin)
#if __has_builtin(__builtin_amdgcn_tensor_load_to_lds)
#define FP_HAS_TDM 1
#endif
#endif

// ---------------------------------------------------------------------------
// bf16 <-> f32 helpers (round-to-nearest-even)
// ---------------------------------------------------------------------------
static __device__ __forceinline__ unsigned short f2bf(float f) {
    union { float f; unsigned u; } x; x.f = f;
    unsigned r = x.u + 0x7FFFu + ((x.u >> 16) & 1u);
    return (unsigned short)(r >> 16);
}
static __device__ __forceinline__ float bf2f(unsigned short h) {
    union { unsigned u; float f; } x; x.u = ((unsigned)h) << 16;
    return x.f;
}

// ---------------------------------------------------------------------------
// Kernel 0: convert fp32 weights -> bf16 (W1: 256x384, W2: 256x256)
// ---------------------------------------------------------------------------
__global__ void fp_convert_w_kernel(const float* __restrict__ W1,
                                    const float* __restrict__ W2,
                                    unsigned short* __restrict__ W1b,
                                    unsigned short* __restrict__ W2b) {
    int i = blockIdx.x * blockDim.x + threadIdx.x;
    const int n1 = HH * KIN1;            // 98304
    const int n2 = HH * HH;              // 65536
    if (i < n1)               W1b[i]       = f2bf(W1[i]);
    else if (i < n1 + n2)     W2b[i - n1]  = f2bf(W2[i - n1]);
}

// ---------------------------------------------------------------------------
// Kernel 1: KNN(k=3) + inverse-distance interpolation + concat -> Xcat
// Xcat layout: (B*N2) rows x 384 bf16 cols (point-major, channel contiguous)
// ---------------------------------------------------------------------------
__global__ __launch_bounds__(256) void fp_knn_interp_kernel(
    const float* __restrict__ xyz1, const float* __restrict__ xyz2,
    const float* __restrict__ f1,   const float* __restrict__ f2,
    unsigned short* __restrict__ Xcat) {
    __shared__ float sx[NN1], sy[NN1], sz[NN1];
    const int b   = blockIdx.y;
    const int n2  = blockIdx.x * 256 + threadIdx.x;

    for (int i = threadIdx.x; i < NN1; i += 256) {
        const float* p = xyz1 + ((size_t)b * NN1 + i) * 3;
        sx[i] = p[0]; sy[i] = p[1]; sz[i] = p[2];
    }
    __syncthreads();

    const float* q = xyz2 + ((size_t)b * NN2 + n2) * 3;
    const float qx = q[0], qy = q[1], qz = q[2];

    float d0 = 3.4e38f, d1 = 3.4e38f, d2 = 3.4e38f;
    int   i0 = 0, i1 = 0, i2 = 0;
    for (int i = 0; i < NN1; ++i) {
        float dx = qx - sx[i], dy = qy - sy[i], dz = qz - sz[i];
        float d  = dx * dx + dy * dy + dz * dz;
        if (d < d0)      { d2 = d1; i2 = i1; d1 = d0; i1 = i0; d0 = d; i0 = i; }
        else if (d < d1) { d2 = d1; i2 = i1; d1 = d;  i1 = i; }
        else if (d < d2) { d2 = d;  i2 = i; }
    }
    float a0 = 1.0f / fmaxf(d0, DIST_MIN);
    float a1 = 1.0f / fmaxf(d1, DIST_MIN);
    float a2 = 1.0f / fmaxf(d2, DIST_MIN);
    float inv = 1.0f / (a0 + a1 + a2);
    float w0 = a0 * inv, w1 = a1 * inv, w2 = a2 * inv;

    unsigned short* row = Xcat + ((size_t)b * NN2 + n2) * KIN1;
    const float* f1b = f1 + (size_t)b * CC1 * NN1;
    for (int c = 0; c < CC1; ++c) {
        const float* fc = f1b + (size_t)c * NN1;
        float v = w0 * fc[i0] + w1 * fc[i1] + w2 * fc[i2];
        row[c] = f2bf(v);
    }
    const float* f2b = f2 + (size_t)b * CC2 * NN2;
    for (int c = 0; c < CC2; ++c)
        row[CC1 + c] = f2bf(f2b[(size_t)c * NN2 + n2]);
}

// ---------------------------------------------------------------------------
// Kernel 2: bf16 WMMA GEMM with double-buffered TDM-staged X panels.
//   Y(n, m) = sum_k W(m,k) * X(n,k) + bias(m)
//   W: 256 x K row-major bf16 (direct global loads, L0/L2 resident)
//   X: NTOT x K row-major bf16. Per K-step a 128-row x 32-k panel is DMA'd
//      into LDS by the Tensor Data Mover; panel p+1 is in flight (TENSORcnt)
//      while the 4 waves run WMMAs on panel p.
//   Y: NTOT x 256 row-major bf16
// Block = 4 waves = 64(M) x 128(N) tile; each wave: 32x64 -> 8 WMMA / K-step.
// ---------------------------------------------------------------------------
__global__ __launch_bounds__(128) void fp_gemm_bf16_kernel(
    const unsigned short* __restrict__ X,
    const unsigned short* __restrict__ W,
    const float* __restrict__ bias,
    unsigned short* __restrict__ Y,
    int K) {
    __shared__ unsigned short sX[2][128 * 32];    // 2 x 8 KB panels

    const int tid  = threadIdx.x;
    const int lane = tid & 31;
    const int wv   = tid >> 5;                    // wave 0..3
    const int g    = lane >> 4;                   // lane group 0/1
    const int l    = lane & 15;
    const int wm   = wv >> 1;                     // M half of block tile
    const int wn   = wv & 1;                      // N half of block tile
    const int mbase = blockIdx.x * 64 + wm * 32;
    const int n0    = blockIdx.y * 128;           // panel base row

    v8f acc[2][4];
    const v8f vzero = {0.f, 0.f, 0.f, 0.f, 0.f, 0.f, 0.f, 0.f};
#pragma unroll
    for (int i = 0; i < 2; ++i)
#pragma unroll
        for (int j = 0; j < 4; ++j) acc[i][j] = vzero;

    const unsigned short* wrow0 = W + (size_t)(mbase + l) * K;
    const unsigned short* wrow1 = W + (size_t)(mbase + 16 + l) * K;

#ifdef FP_HAS_TDM
    // Tensor DMA descriptor group 1 (256b): data_size=2B, tensor K x NTOT,
    // tile 32(k) x 128(n), dim0 stride = K elements. Groups 2/3 = 0 (2D tile).
    const unsigned long long q0 = (1ull << 16) |
                                  ((unsigned long long)((unsigned)K & 0xFFFFu) << 48);
    const unsigned long long q1 = (unsigned long long)(((unsigned)K >> 16) & 0xFFFFu) |
                                  ((unsigned long long)(unsigned)NTOT << 16) |
                                  (32ull << 48);
    const unsigned long long q2 = 128ull | ((unsigned long long)(unsigned)K << 32);
    const i32x8 g1v = {(int)(unsigned)q0, (int)(q0 >> 32),
                       (int)(unsigned)q1, (int)(q1 >> 32),
                       (int)(unsigned)q2, (int)(q2 >> 32), 0, 0};
    const i32x4 gzv = {0, 0, 0, 0};
    const unsigned lds0 = (unsigned)(unsigned long long)(uintptr_t)(&sX[0][0]);
    const unsigned lds1 = (unsigned)(unsigned long long)(uintptr_t)(&sX[1][0]);
    const unsigned long long xbase =
        (unsigned long long)(uintptr_t)X + (unsigned long long)n0 * K * 2ull;

    // Prologue: kick off panel 0 into buffer 0.
    if (wv == 0) {
        u32x4 g0v = {1u, lds0, (unsigned)xbase,
                     (unsigned)((xbase >> 32) & 0x1FFFFFFu) | (2u << 30)};
#if defined(__clang_major__) && (__clang_major__ >= 23)
        const i32x8 z8 = {0, 0, 0, 0, 0, 0, 0, 0};
        __builtin_amdgcn_tensor_load_to_lds(g0v, g1v, gzv, gzv, z8, 0);
#else
        __builtin_amdgcn_tensor_load_to_lds(g0v, g1v, gzv, gzv, 0);
#endif
    }
#endif

    for (int kb = 0; kb < K; kb += 32) {
        const int buf = (kb >> 5) & 1;
#ifdef FP_HAS_TDM
        if (wv == 0) __builtin_amdgcn_s_wait_tensorcnt(0);  // panel kb arrived
        __syncthreads();   // publish panel kb; also: no wave still reads buf^1
        if (wv == 0 && (kb + 32) < K) {
            unsigned long long ga = xbase + (unsigned long long)(kb + 32) * 2ull;
            u32x4 g0v = {1u, buf ? lds0 : lds1, (unsigned)ga,
                         (unsigned)((ga >> 32) & 0x1FFFFFFu) | (2u << 30)};
#if defined(__clang_major__) && (__clang_major__ >= 23)
            const i32x8 z8 = {0, 0, 0, 0, 0, 0, 0, 0};
            __builtin_amdgcn_tensor_load_to_lds(g0v, g1v, gzv, gzv, z8, 0);
#else
            __builtin_amdgcn_tensor_load_to_lds(g0v, g1v, gzv, gzv, 0);
#endif
        }
#else
        // Fallback: cooperative copy of the 128x32 panel into LDS.
        __syncthreads();
        for (int e = tid * 8; e < 128 * 32; e += 128 * 8) {
            const int row = e >> 5, col = e & 31;
            *(uint4*)&sX[buf][e] =
                *(const uint4*)&X[(size_t)(n0 + row) * K + kb + col];
        }
        __syncthreads();
#endif

        v16bf afrag[2];
#pragma unroll
        for (int i = 0; i < 2; ++i) {
            const unsigned short* wr = (i == 0) ? wrow0 : wrow1;
            v8bf lo = *reinterpret_cast<const v8bf*>((const void*)(wr + kb + g * 16));
            v8bf hi = *reinterpret_cast<const v8bf*>((const void*)(wr + kb + g * 16 + 8));
            afrag[i] = __builtin_shufflevector(lo, hi, 0, 1, 2, 3, 4, 5, 6, 7,
                                                        8, 9, 10, 11, 12, 13, 14, 15);
        }
        v16bf bfrag[4];
#pragma unroll
        for (int j = 0; j < 4; ++j) {
            const int prow = wn * 64 + j * 16 + l;
            bfrag[j] = *reinterpret_cast<const v16bf*>(
                (const void*)&sX[buf][prow * 32 + g * 16]);
        }
#pragma unroll
        for (int i = 0; i < 2; ++i)
#pragma unroll
            for (int j = 0; j < 4; ++j)
                acc[i][j] = __builtin_amdgcn_wmma_f32_16x16x32_bf16(
                    false, afrag[i], false, bfrag[j], (short)0, acc[i][j],
                    false, false);
    }

    // Epilogue: +bias, convert to bf16, store point-major rows of Y.
    // C/D layout: VGPR v, lanes 0-15: M=v ; lanes 16-31: M=v+8 ; N=lane%16.
#pragma unroll
    for (int i = 0; i < 2; ++i) {
        const int mcol = mbase + i * 16 + g * 8;   // 8 contiguous channels
        float bsv[8];
#pragma unroll
        for (int v = 0; v < 8; ++v) bsv[v] = bias[mcol + v];
#pragma unroll
        for (int j = 0; j < 4; ++j) {
            const int n = n0 + wn * 64 + j * 16 + l;
            v8us pk;
#pragma unroll
            for (int v = 0; v < 8; ++v) pk[v] = f2bf(acc[i][j][v] + bsv[v]);
            *reinterpret_cast<v8us*>((void*)(Y + (size_t)n * HH + mcol)) = pk;
        }
    }
}

// ---------------------------------------------------------------------------
// Kernel 3: per-channel BN stats over (B,N): fold into scale/shift
//   ss[m] = gamma[m]*rsqrt(var+eps), ss[256+m] = beta[m]-mean*scale
// ---------------------------------------------------------------------------
__global__ __launch_bounds__(256) void fp_bn_stats_kernel(
    const unsigned short* __restrict__ Y,
    const float* __restrict__ gamma, const float* __restrict__ beta,
    float* __restrict__ ss) {
    __shared__ float rs[256], rq[256];
    const int m = blockIdx.x;
    float s = 0.f, q = 0.f;
    for (int n = threadIdx.x; n < NTOT; n += 256) {
        float f = bf2f(Y[(size_t)n * HH + m]);
        s += f; q += f * f;
    }
    rs[threadIdx.x] = s; rq[threadIdx.x] = q;
    __syncthreads();
    for (int off = 128; off > 0; off >>= 1) {
        if (threadIdx.x < off) {
            rs[threadIdx.x] += rs[threadIdx.x + off];
            rq[threadIdx.x] += rq[threadIdx.x + off];
        }
        __syncthreads();
    }
    if (threadIdx.x == 0) {
        float mean = rs[0] * (1.0f / NTOT);
        float var  = rq[0] * (1.0f / NTOT) - mean * mean;
        float scl  = gamma[m] * rsqrtf(var + BN_EPS);
        ss[m]       = scl;
        ss[HH + m]  = beta[m] - mean * scl;
    }
}

// ---------------------------------------------------------------------------
// Kernel 4: elementwise BN + ReLU -> bf16 activations (GEMM2 input)
// ---------------------------------------------------------------------------
__global__ __launch_bounds__(256) void fp_bn_relu_kernel(
    const unsigned short* __restrict__ Y, const float* __restrict__ ss,
    unsigned short* __restrict__ Xo) {
    size_t idx = (size_t)blockIdx.x * 256 + threadIdx.x;
    int m = (int)(idx & (HH - 1));
    float f = bf2f(Y[idx]) * ss[m] + ss[HH + m];
    Xo[idx] = f2bf(fmaxf(f, 0.f));
}

// ---------------------------------------------------------------------------
// Kernel 5: final BN + ReLU, transpose-write fp32 output (B, H, N2)
// ---------------------------------------------------------------------------
__global__ __launch_bounds__(256) void fp_final_kernel(
    const unsigned short* __restrict__ Y, const float* __restrict__ ss,
    float* __restrict__ out) {
    size_t idx = (size_t)blockIdx.x * 256 + threadIdx.x;   // over B*H*N2
    int n2 = (int)(idx % NN2);
    int m  = (int)((idx / NN2) % HH);
    int b  = (int)(idx / ((size_t)NN2 * HH));
    float f = bf2f(Y[((size_t)b * NN2 + n2) * HH + m]) * ss[m] + ss[HH + m];
    out[idx] = fmaxf(f, 0.f);
}

// ---------------------------------------------------------------------------
// Host-side launch
// ---------------------------------------------------------------------------
extern "C" void kernel_launch(void* const* d_in, const int* in_sizes, int n_in,
                              void* d_out, int out_size, void* d_ws, size_t ws_size,
                              hipStream_t stream) {
    const float* xyz1   = (const float*)d_in[0];
    const float* xyz2   = (const float*)d_in[1];
    const float* feat1  = (const float*)d_in[2];
    const float* feat2  = (const float*)d_in[3];
    const float* W1     = (const float*)d_in[4];
    const float* b1     = (const float*)d_in[5];
    const float* gamma1 = (const float*)d_in[6];
    const float* beta1  = (const float*)d_in[7];
    const float* W2     = (const float*)d_in[8];
    const float* b2     = (const float*)d_in[9];
    const float* gamma2 = (const float*)d_in[10];
    const float* beta2  = (const float*)d_in[11];
    float* out = (float*)d_out;

    // Workspace layout (bytes). Total ~84.2 MB.
    char* ws = (char*)d_ws;
    const size_t OFF_W1B  = 0;                                  // 196608
    const size_t OFF_W2B  = OFF_W1B + (size_t)HH * KIN1 * 2;    // +131072
    const size_t OFF_SS1  = OFF_W2B + (size_t)HH * HH * 2;      // +2048
    const size_t OFF_SS2  = OFF_SS1 + 2 * HH * sizeof(float);   // +2048
    const size_t OFF_XCAT = 331776;                             // 256-aligned
    const size_t OFF_Y    = OFF_XCAT + (size_t)NTOT * KIN1 * 2; // 50,663,424

    unsigned short* W1b  = (unsigned short*)(ws + OFF_W1B);
    unsigned short* W2b  = (unsigned short*)(ws + OFF_W2B);
    float*          ss1  = (float*)(ws + OFF_SS1);
    float*          ss2  = (float*)(ws + OFF_SS2);
    unsigned short* Xcat = (unsigned short*)(ws + OFF_XCAT);
    unsigned short* Ybuf = (unsigned short*)(ws + OFF_Y);
    unsigned short* Xact = Xcat;   // reuse: Xcat dead after GEMM1
    unsigned short* Y2   = Ybuf;   // reuse: Y1pre dead after bn_relu

    // 0) weights -> bf16
    {
        int tot = HH * KIN1 + HH * HH;
        fp_convert_w_kernel<<<(tot + 255) / 256, 256, 0, stream>>>(W1, W2, W1b, W2b);
    }
    // 1) KNN + interpolate + concat
    fp_knn_interp_kernel<<<dim3(NN2 / 256, BB), 256, 0, stream>>>(
        xyz1, xyz2, feat1, feat2, Xcat);
    // 2) GEMM1 (K=384) -> Y1pre
    fp_gemm_bf16_kernel<<<dim3(HH / 64, NTOT / 128), 128, 0, stream>>>(
        Xcat, W1b, b1, Ybuf, KIN1);
    // 3) BN1 stats -> scale/shift
    fp_bn_stats_kernel<<<HH, 256, 0, stream>>>(Ybuf, gamma1, beta1, ss1);
    // 4) BN1 + ReLU -> activations
    fp_bn_relu_kernel<<<(unsigned)((size_t)NTOT * HH / 256), 256, 0, stream>>>(
        Ybuf, ss1, Xact);
    // 5) GEMM2 (K=256) -> Y2pre
    fp_gemm_bf16_kernel<<<dim3(HH / 64, NTOT / 128), 128, 0, stream>>>(
        Xact, W2b, b2, Y2, HH);
    // 6) BN2 stats
    fp_bn_stats_kernel<<<HH, 256, 0, stream>>>(Y2, gamma2, beta2, ss2);
    // 7) BN2 + ReLU + transpose to (B, H, N2) fp32
    fp_final_kernel<<<(unsigned)((size_t)BB * HH * NN2 / 256), 256, 0, stream>>>(
        Y2, ss2, out);
}
